// MambaBackbone_60146722013732
// MI455X (gfx1250) — matrix operand order repaired and checked by reading, not compiled, not run on hardware
//
#include <hip/hip_runtime.h>
#include <math.h>

// ---------------------------------------------------------------------------
// Mamba backbone forward, MI455X (gfx1250, wave32, WMMA).
// All fp32->bf16(hi,lo) splitting is done ONCE by producers (LN / conv / GEMM
// epilogues / weight-prep), so the GEMM hot loop is pure b128 loads + 12
// v_wmma_f32_16x16x32_bf16 per K-step (3-term bf16 split x 4 N-tiles),
// issued term-major so adjacent WMMAs hit different accumulators (hides the
// 5-NOP WMMA->WMMA RAW hazard).
// ---------------------------------------------------------------------------

#define D_MODEL  512
#define D_INNER  1024
#define DT_RANK  32
#define D_STATE  16
#define D_CONV   4
#define NBATCH   4
#define SEQLEN   1024
#define NTOK     (NBATCH * SEQLEN)        // 4096
#define NLAYERS  4
#define XZCOLS   (2 * D_INNER)            // 2048
#define DBLCOLS  (DT_RANK + 2 * D_STATE)  // 64

typedef __attribute__((ext_vector_type(16))) __bf16 v16bf;
typedef __attribute__((ext_vector_type(8)))  __bf16 v8bf;
typedef __attribute__((ext_vector_type(8)))  float  v8f;

__device__ __forceinline__ float sig_f(float x) { return 1.0f / (1.0f + expf(-x)); }

__device__ __forceinline__ void split_bf16(float x, __bf16& hi, __bf16& lo) {
    hi = (__bf16)x;                 // RNE to bf16
    lo = (__bf16)(x - (float)hi);   // residual term
}

__device__ __forceinline__ v16bf cat8(v8bf a, v8bf b) {
    return __builtin_shufflevector(a, b, 0, 1, 2, 3, 4, 5, 6, 7,
                                         8, 9, 10, 11, 12, 13, 14, 15);
}

// ---------------------------------------------------------------------------
// Elementwise fp32 -> bf16 hi/lo planes (weight prep).
// ---------------------------------------------------------------------------
__global__ void __launch_bounds__(256) split_kernel(
    const float* __restrict__ in, __bf16* __restrict__ hi,
    __bf16* __restrict__ lo, int n)
{
    const int idx = blockIdx.x * 256 + threadIdx.x;
    if (idx >= n) return;
    __bf16 h, l;
    split_bf16(in[idx], h, l);
    hi[idx] = h;
    lo[idx] = l;
}

// ---------------------------------------------------------------------------
// GEMM: C[M,N] = Ahl[M,K] * Whl[N,K]^T  (+resid), bf16x3 with f32 accumulate.
// A/W given as bf16 hi/lo planes (row-major). One wave -> 16(M) x 64(N) strip.
// Optional epilogue: f32 C and/or bf16 hi/lo C planes.
// M%16==0, N%64==0, K%32==0.
// ---------------------------------------------------------------------------
__global__ void __launch_bounds__(256) gemm_bf16x3_kernel(
    const __bf16* __restrict__ Ahi, const __bf16* __restrict__ Alo, int lda,
    const __bf16* __restrict__ Whi, const __bf16* __restrict__ Wlo, int ldw,
    const float* __restrict__ resid,
    float* __restrict__ C, __bf16* __restrict__ Chi, __bf16* __restrict__ Clo,
    int ldc, int M, int N, int K)
{
    const int lane = threadIdx.x & 31;
    const int wid  = blockIdx.x * 8 + (threadIdx.x >> 5);
    const int ngroups = N >> 6;
    const int mtiles  = M >> 4;
    if (wid >= mtiles * ngroups) return;   // wave-uniform
    const int mt = wid / ngroups;
    const int ng = wid % ngroups;

    const int half = lane >> 4;   // 0: lanes 0-15, 1: lanes 16-31
    const int l16  = lane & 15;

    v8f acc[4] = {};

    const __bf16* ArowH = Ahi + (size_t)(mt * 16 + l16) * lda;
    const __bf16* ArowL = Alo + (size_t)(mt * 16 + l16) * lda;

    for (int k0 = 0; k0 < K; k0 += 32) {
        // A fragment (16x32): element e -> K = 16*(e/8) + 8*half + e%8
        // -> two contiguous 8-bf16 (16 B) runs per plane.
        v8bf ah0 = *(const v8bf*)(ArowH + k0 + 8 * half);
        v8bf ah1 = *(const v8bf*)(ArowH + k0 + 16 + 8 * half);
        v8bf al0 = *(const v8bf*)(ArowL + k0 + 8 * half);
        v8bf al1 = *(const v8bf*)(ArowL + k0 + 16 + 8 * half);
        v16bf afh = cat8(ah0, ah1);
        v16bf afl = cat8(al0, al1);

        // B fragments (32x16): lane n = l16, element e -> K = k0 + 16*(e/8) + 8*half + e%8
        // -> two contiguous 8-bf16 (16 B) runs per plane.
        v16bf bfh[4], bfl[4];
        #pragma unroll
        for (int nt = 0; nt < 4; ++nt) {
            const int n = ng * 64 + nt * 16 + l16;
            const __bf16* WrowH = Whi + (size_t)n * ldw + k0 + 8 * half;
            const __bf16* WrowL = Wlo + (size_t)n * ldw + k0 + 8 * half;
            bfh[nt] = cat8(*(const v8bf*)(WrowH), *(const v8bf*)(WrowH + 16));
            bfl[nt] = cat8(*(const v8bf*)(WrowL), *(const v8bf*)(WrowL + 16));
        }

        // Term-major issue: adjacent WMMAs use different accumulators.
        #pragma unroll
        for (int nt = 0; nt < 4; ++nt)
            acc[nt] = __builtin_amdgcn_wmma_f32_16x16x32_bf16(false, afh, false, bfh[nt], (short)0, acc[nt], false, false);
        #pragma unroll
        for (int nt = 0; nt < 4; ++nt)
            acc[nt] = __builtin_amdgcn_wmma_f32_16x16x32_bf16(false, afh, false, bfl[nt], (short)0, acc[nt], false, false);
        #pragma unroll
        for (int nt = 0; nt < 4; ++nt)
            acc[nt] = __builtin_amdgcn_wmma_f32_16x16x32_bf16(false, afl, false, bfh[nt], (short)0, acc[nt], false, false);
    }

    // Store: D row M = v + 8*half, col N = l16.
    #pragma unroll
    for (int nt = 0; nt < 4; ++nt) {
        const int n = ng * 64 + nt * 16 + l16;
        #pragma unroll
        for (int v = 0; v < 8; ++v) {
            const int m = mt * 16 + v + 8 * half;
            const size_t off = (size_t)m * ldc + n;
            float val = acc[nt][v];
            if (resid) val += resid[off];
            if (C) C[off] = val;
            if (Chi) { __bf16 h, l; split_bf16(val, h, l); Chi[off] = h; Clo[off] = l; }
        }
    }
}

// ---------------------------------------------------------------------------
// LayerNorm over rows of 512 -> bf16 hi/lo planes.  One block per row.
// ---------------------------------------------------------------------------
__global__ void __launch_bounds__(256) layernorm_kernel(
    const float* __restrict__ x, const float* __restrict__ g,
    const float* __restrict__ b, __bf16* __restrict__ ohi,
    __bf16* __restrict__ olo)
{
    __shared__ float s1[256], s2[256];
    const int row = blockIdx.x;
    const int tid = threadIdx.x;
    const float* xr = x + (size_t)row * D_MODEL;
    float v0 = xr[tid], v1 = xr[tid + 256];
    s1[tid] = v0 + v1;
    s2[tid] = v0 * v0 + v1 * v1;
    __syncthreads();
    for (int s = 128; s > 0; s >>= 1) {
        if (tid < s) { s1[tid] += s1[tid + s]; s2[tid] += s2[tid + s]; }
        __syncthreads();
    }
    const float mean = s1[0] * (1.0f / D_MODEL);
    const float var  = s2[0] * (1.0f / D_MODEL) - mean * mean;
    const float inv  = rsqrtf(var + 1e-5f);
    const size_t o0 = (size_t)row * D_MODEL + tid;
    const size_t o1 = o0 + 256;
    __bf16 h, l;
    split_bf16((v0 - mean) * inv * g[tid] + b[tid], h, l);
    ohi[o0] = h; olo[o0] = l;
    split_bf16((v1 - mean) * inv * g[tid + 256] + b[tid + 256], h, l);
    ohi[o1] = h; olo[o1] = l;
}

// ---------------------------------------------------------------------------
// Depthwise causal conv(4) + bias + SiLU.  Reads xc half of xz (f32),
// writes f32 (for scan) and bf16 hi/lo (for x-proj GEMM).
// ---------------------------------------------------------------------------
__global__ void __launch_bounds__(256) conv_silu_kernel(
    const float* __restrict__ xz, const float* __restrict__ cw,
    const float* __restrict__ cb, float* __restrict__ xc,
    __bf16* __restrict__ xchi, __bf16* __restrict__ xclo)
{
    const int idx = blockIdx.x * 256 + threadIdx.x;   // NTOK*D_INNER total
    const int d = idx & (D_INNER - 1);
    const int l = (idx >> 10) & (SEQLEN - 1);
    const int b = idx >> 20;
    const float* col = xz + (size_t)b * SEQLEN * XZCOLS + d;
    float acc = cb[d];
    #pragma unroll
    for (int j = 0; j < D_CONV; ++j) {
        const int t = l - (D_CONV - 1) + j;
        if (t >= 0) acc += cw[d * D_CONV + j] * col[(size_t)t * XZCOLS];
    }
    const float s = acc * sig_f(acc);   // SiLU
    xc[idx] = s;
    __bf16 h, lo;
    split_bf16(s, h, lo);
    xchi[idx] = h; xclo[idx] = lo;
}

// ---------------------------------------------------------------------------
// Selective scan.  One lane per (batch, channel) chain; 16-wide state in
// registers.  Fuses softplus(dt), dB*x, C-contraction, D-skip, z-gate.
// Writes y directly as bf16 hi/lo planes for the out-proj GEMM.
// ---------------------------------------------------------------------------
__global__ void __launch_bounds__(32) scan_kernel(
    const float* __restrict__ dtraw, const float* __restrict__ dt_b,
    const float* __restrict__ A_log, const float* __restrict__ Dp,
    const float* __restrict__ xc, const float* __restrict__ xz,
    const float* __restrict__ dbl, __bf16* __restrict__ yhi,
    __bf16* __restrict__ ylo)
{
    const int d = (blockIdx.x & 31) * 32 + threadIdx.x;  // 0..1023
    const int b = blockIdx.x >> 5;                       // 0..3
    float Aq[D_STATE];
    #pragma unroll
    for (int s = 0; s < D_STATE; ++s) Aq[s] = -expf(A_log[d * D_STATE + s]);
    const float dpd = Dp[d];
    const float dtb = dt_b[d];
    float h[D_STATE] = {};

    for (int l = 0; l < SEQLEN; ++l) {
        const size_t tok = (size_t)b * SEQLEN + l;
        const float dr = dtraw[tok * D_INNER + d] + dtb;
        const float dt = (dr > 20.0f) ? dr : log1pf(expf(dr));   // softplus
        const float xcv = xc[tok * D_INNER + d];
        const float zv  = xz[tok * XZCOLS + D_INNER + d];
        const float* bc = dbl + tok * DBLCOLS;
        const float dtx = dt * xcv;
        float yv = 0.0f;
        #pragma unroll
        for (int s = 0; s < D_STATE; ++s) {
            const float dA = expf(dt * Aq[s]);
            h[s] = dA * h[s] + dtx * bc[DT_RANK + s];            // B
            yv  += h[s] * bc[DT_RANK + D_STATE + s];             // C
        }
        yv += dpd * xcv;
        yv *= zv * sig_f(zv);                                    // z-gate
        __bf16 hh, ll;
        split_bf16(yv, hh, ll);
        yhi[tok * D_INNER + d] = hh;
        ylo[tok * D_INNER + d] = ll;
    }
}

// ---------------------------------------------------------------------------
static inline void launch_gemm(const __bf16* Ahi, const __bf16* Alo, int lda,
                               const __bf16* Whi, const __bf16* Wlo, int ldw,
                               const float* resid, float* C,
                               __bf16* Chi, __bf16* Clo, int ldc,
                               int M, int N, int K, hipStream_t s)
{
    const int waves  = (M / 16) * (N / 64);
    const int blocks = (waves + 7) / 8;
    gemm_bf16x3_kernel<<<blocks, 256, 0, s>>>(Ahi, Alo, lda, Whi, Wlo, ldw,
                                              resid, C, Chi, Clo, ldc, M, N, K);
}

static inline void launch_split(const float* in, __bf16* hi, __bf16* lo,
                                int n, hipStream_t s)
{
    split_kernel<<<(n + 255) / 256, 256, 0, s>>>(in, hi, lo, n);
}

extern "C" void kernel_launch(void* const* d_in, const int* in_sizes, int n_in,
                              void* d_out, int out_size, void* d_ws, size_t ws_size,
                              hipStream_t stream)
{
    (void)in_sizes; (void)n_in; (void)out_size; (void)ws_size;

    const float* x_in   = (const float*)d_in[0];
    const float* ln_g   = (const float*)d_in[1];
    const float* ln_b   = (const float*)d_in[2];
    const float* in_w   = (const float*)d_in[3];
    const float* conv_w = (const float*)d_in[4];
    const float* conv_b = (const float*)d_in[5];
    const float* xp_w   = (const float*)d_in[6];
    const float* dt_w   = (const float*)d_in[7];
    const float* dt_b   = (const float*)d_in[8];
    const float* A_log  = (const float*)d_in[9];
    const float* Dp     = (const float*)d_in[10];
    const float* out_w  = (const float*)d_in[11];

    float* x = (float*)d_out;  // running residual stream [NTOK, 512]
    hipMemcpyAsync(x, x_in, (size_t)NTOK * D_MODEL * sizeof(float),
                   hipMemcpyDeviceToDevice, stream);

    // --- workspace layout (byte offsets; ~105 MB total) -------------------
    char* base = (char*)d_ws;
    const size_t MBy = 1024 * 1024;
    float*  xzbuf  = (float*)(base + 0 * MBy);     // [NTOK,2048] f32, 32MB
    float*  xcbuf  = (float*)(base + 32 * MBy);    // [NTOK,1024] f32, 16MB
    float*  dtbuf  = (float*)(base + 48 * MBy);    // [NTOK,1024] f32, 16MB
    float*  dblf   = (float*)(base + 64 * MBy);    // [NTOK,  64] f32,  1MB
    __bf16* xch    = (__bf16*)(base + 65 * MBy);   // [NTOK,1024] bf16, 8MB
    __bf16* xcl    = (__bf16*)(base + 73 * MBy);   //                  8MB
    __bf16* dblh   = (__bf16*)(base + 81 * MBy);   // [NTOK,  64] bf16, .5MB
    __bf16* dbll   = (__bf16*)(base + 81 * MBy + 512 * 1024);
    __bf16* yh     = (__bf16*)(base + 82 * MBy);   // [NTOK,1024] bf16, 8MB
    __bf16* yl     = (__bf16*)(base + 90 * MBy);   //                  8MB
    // per-layer weight planes (reused each layer)
    __bf16* wih    = (__bf16*)(base + 98 * MBy);   // in_w  1M elems, 2MB
    __bf16* wil    = (__bf16*)(base + 100 * MBy);
    __bf16* woh    = (__bf16*)(base + 102 * MBy);  // out_w 512K elems, 1MB
    __bf16* wol    = (__bf16*)(base + 103 * MBy);
    __bf16* wxh    = (__bf16*)(base + 104 * MBy);  // xp_w 64K elems, 128KB
    __bf16* wxl    = (__bf16*)(base + 104 * MBy + 128 * 1024);
    __bf16* wdh    = (__bf16*)(base + 104 * MBy + 256 * 1024);  // dt_w 32K
    __bf16* wdl    = (__bf16*)(base + 104 * MBy + 320 * 1024);
    // xln bf16 planes alias dtbuf (dead before dt GEMM writes it)
    __bf16* xlnh   = (__bf16*)(base + 48 * MBy);   // [NTOK,512] bf16, 4MB
    __bf16* xlnl   = (__bf16*)(base + 52 * MBy);   //                  4MB

    for (int i = 0; i < NLAYERS; ++i) {
        const float* in_w_i   = in_w   + (size_t)i * XZCOLS * D_MODEL;
        const float* conv_w_i = conv_w + (size_t)i * D_INNER * D_CONV;
        const float* conv_b_i = conv_b + (size_t)i * D_INNER;
        const float* xp_w_i   = xp_w   + (size_t)i * DBLCOLS * D_INNER;
        const float* dt_w_i   = dt_w   + (size_t)i * D_INNER * DT_RANK;
        const float* dt_b_i   = dt_b   + (size_t)i * D_INNER;
        const float* A_log_i  = A_log  + (size_t)i * D_INNER * D_STATE;
        const float* Dp_i     = Dp     + (size_t)i * D_INNER;
        const float* out_w_i  = out_w  + (size_t)i * D_MODEL * D_INNER;

        // 0) weight prep: fp32 -> bf16 hi/lo planes (once per layer)
        launch_split(in_w_i,  wih, wil, XZCOLS * D_MODEL,  stream);
        launch_split(xp_w_i,  wxh, wxl, DBLCOLS * D_INNER, stream);
        launch_split(dt_w_i,  wdh, wdl, D_INNER * DT_RANK, stream);
        launch_split(out_w_i, woh, wol, D_MODEL * D_INNER, stream);

        // 1) LayerNorm -> xln bf16 hi/lo
        layernorm_kernel<<<NTOK, 256, 0, stream>>>(x, ln_g + i * D_MODEL,
                                                   ln_b + i * D_MODEL,
                                                   xlnh, xlnl);
        // 2) xz = xln @ in_w^T            (4096 x 2048 x 512), f32 out
        launch_gemm(xlnh, xlnl, D_MODEL, wih, wil, D_MODEL, nullptr,
                    xzbuf, nullptr, nullptr, XZCOLS, NTOK, XZCOLS, D_MODEL,
                    stream);
        // 3) depthwise causal conv + SiLU -> xc f32 + bf16 hi/lo
        conv_silu_kernel<<<(NTOK * D_INNER) / 256, 256, 0, stream>>>(
            xzbuf, conv_w_i, conv_b_i, xcbuf, xch, xcl);
        // 4) dbl = xc @ xp_w^T            (4096 x 64 x 1024), f32 + bf16 out
        launch_gemm(xch, xcl, D_INNER, wxh, wxl, D_INNER, nullptr,
                    dblf, dblh, dbll, DBLCOLS, NTOK, DBLCOLS, D_INNER, stream);
        // 5) dtraw = dbl[:, :32] @ dt_w^T (4096 x 1024 x 32), f32 out
        launch_gemm(dblh, dbll, DBLCOLS, wdh, wdl, DT_RANK, nullptr,
                    dtbuf, nullptr, nullptr, D_INNER, NTOK, D_INNER, DT_RANK,
                    stream);
        // 6) selective scan -> y bf16 hi/lo
        scan_kernel<<<NBATCH * (D_INNER / 32), 32, 0, stream>>>(
            dtbuf, dt_b_i, A_log_i, Dp_i, xcbuf, xzbuf, dblf, yh, yl);
        // 7) x = x + y @ out_w^T          (4096 x 512 x 1024), fused residual
        launch_gemm(yh, yl, D_INNER, woh, wol, D_INNER, /*resid=*/x,
                    x, nullptr, nullptr, D_MODEL, NTOK, D_MODEL, D_INNER,
                    stream);
    }
}
